// Self_Attention_29360396436110
// MI455X (gfx1250) — matrix operand order repaired and checked by reading, not compiled, not run on hardware
//
#include <hip/hip_runtime.h>

// ---------------------------------------------------------------------------
// Self-attention for B=8, S=2048, D=1024 (single head), fp32 in/out.
// All matmuls run on v_wmma_f32_16x16x32_bf16 (wave32, CDNA5), with
// double-buffered GLOBAL_LOAD_ASYNC_TO_LDS_B128 staging (ASYNCcnt pipeline).
// Pipeline:
//   1. convert x -> bf16 (xb), convert+transpose weights -> wbT[e][d] bf16
//   2. Q = xb @ Wq, K = xb @ Wk (bf16, row major); V stored transposed Vt[e][m]
//   3. Sc = Q @ K^T (bf16, per batch)
//   4. softmax rows of Sc in place (f32 math, scale 1/sqrt(1024))
//   5. out = Sc @ V   (f32 row-major, via Vt)
// ---------------------------------------------------------------------------

typedef unsigned short u16;
typedef __attribute__((ext_vector_type(16))) __bf16 v16bf;
typedef __attribute__((ext_vector_type(8)))  float  v8f;
typedef int v4i __attribute__((vector_size(16)));   // matches builtin param type

#if defined(__gfx1250__) && \
    __has_builtin(__builtin_amdgcn_global_load_async_to_lds_b128) && \
    __has_builtin(__builtin_amdgcn_s_wait_asynccnt)
#define USE_ASYNC_LDS 1
typedef __attribute__((address_space(1))) v4i* gv4i_p;   // global v4i*
typedef __attribute__((address_space(3))) v4i* lv4i_p;   // LDS v4i*
#else
#define USE_ASYNC_LDS 0
#endif

union FragBF { v16bf v; uint4 q[2]; };

__device__ __forceinline__ u16 f2bf(float f) {
    unsigned u = __float_as_uint(f);
    u += 0x7FFFu + ((u >> 16) & 1u);           // round-to-nearest-even
    return (u16)(u >> 16);
}
__device__ __forceinline__ float bf2f(u16 h) {
    return __uint_as_float((unsigned)h << 16);
}

// ---------------------------------------------------------------------------
// f32 -> bf16 bulk convert (8 elements / thread, b128 store)
// ---------------------------------------------------------------------------
__global__ void conv_f32_to_bf16(const float* __restrict__ in, u16* __restrict__ out) {
    size_t i = ((size_t)blockIdx.x * blockDim.x + threadIdx.x) * 8;
    float4 a = *(const float4*)(in + i);
    float4 b = *(const float4*)(in + i + 4);
    union { uint4 q; u16 h[8]; } u;
    u.h[0] = f2bf(a.x); u.h[1] = f2bf(a.y); u.h[2] = f2bf(a.z); u.h[3] = f2bf(a.w);
    u.h[4] = f2bf(b.x); u.h[5] = f2bf(b.y); u.h[6] = f2bf(b.z); u.h[7] = f2bf(b.w);
    *(uint4*)(out + i) = u.q;
}

// ---------------------------------------------------------------------------
// weight convert + transpose: wbT[w][e][d] = bf16(kernel[w][d][e]); 1024x1024 x3
// 32x32 LDS tile transpose, block (32,8), grid (32,32,3)
// ---------------------------------------------------------------------------
__global__ void conv_transpose_w(const float* __restrict__ in, u16* __restrict__ out) {
    __shared__ float t[32][33];
    const int w  = blockIdx.z;
    const float* src = in  + (size_t)w * 1024 * 1024;
    u16*         dst = out + (size_t)w * 1024 * 1024;
    const int x0 = blockIdx.x * 32, y0 = blockIdx.y * 32;
    const int tx = threadIdx.x, ty = threadIdx.y;
#pragma unroll
    for (int r = 0; r < 4; ++r)
        t[ty + r * 8][tx] = src[(size_t)(y0 + ty + r * 8) * 1024 + (x0 + tx)];
    __syncthreads();
#pragma unroll
    for (int r = 0; r < 4; ++r)
        dst[(size_t)(x0 + ty + r * 8) * 1024 + (y0 + tx)] = f2bf(t[tx][ty + r * 8]);
}

// ---------------------------------------------------------------------------
// LDS fragment loaders. Tiles are row-major [128 rows][64 bf16] (128 B rows).
// A frag (16x32, M x K): lane r=lane&15 is row; VGPR halves follow the ISA
//   layout: lanes 0-15 hold K {0..7,16..23}, lanes 16-31 hold K {8..15,24..31}.
// B frag (32x16, K x N): lane c=lane&15 is column n; same K layout as A:
//   lanes 0-15 hold K {0..7,16..23}, lanes 16-31 hold K {8..15,24..31}.
// ---------------------------------------------------------------------------
__device__ __forceinline__ v16bf ldfragA(const u16* lds, int mrow, int kf, int lane) {
    const int r = lane & 15, hi = lane >> 4;
    FragBF u;
    const u16* p = lds + (mrow + r) * 64 + kf + hi * 8;
    u.q[0] = *(const uint4*)(p);
    u.q[1] = *(const uint4*)(p + 16);
    return u.v;
}
__device__ __forceinline__ v16bf ldfragB(const u16* lds, int nrow, int kf, int lane) {
    const int c = lane & 15, hi = lane >> 4;
    FragBF u;
    const u16* p = lds + (nrow + c) * 64 + kf + hi * 8;
    u.q[0] = *(const uint4*)(p);
    u.q[1] = *(const uint4*)(p + 16);
    return u.v;
}

// ---------------------------------------------------------------------------
// Generic bf16 WMMA GEMM:  C = A (MxK, row-major, lda) * Bt^T
// where Bt is the transposed right operand stored row-major [N][K] (ldb).
// 128x128 tile / workgroup (256 threads = 8 waves, each wave 32x64).
// K staged 64-wide through LDS; double-buffered via async-to-LDS when present.
// MODE: 0 = bf16 row-major C (ldc); 1 = bf16 transposed C[n*ldc+m]; 2 = f32 C.
// blockIdx.z selects batch via element strides aBS/bBS/cBS.
// M, N multiples of 128; Ktot multiple of 64 (>= 128); pointers 16B aligned.
// __launch_bounds__(256, 1): one 8-wave block per WGP is enough (async DMA
// hides global latency); give the allocator the full VGPR budget -> no spills.
// ---------------------------------------------------------------------------
template <int MODE>
__global__ __launch_bounds__(256, 1)
void gemm_bf16_wmma(const u16* __restrict__ A, long long aBS, int lda,
                    const u16* __restrict__ Bt, long long bBS, int ldb,
                    void* __restrict__ C, long long cBS, int ldc,
                    int Ktot) {
#if USE_ASYNC_LDS
    __shared__ uint4 lbuf[2][2048];   // per buffer: A = [0..1023], B = [1024..2047]; 64 KB
#else
    __shared__ uint4 lbuf[1][2048];   // 32 KB single buffer
#endif

    const int tid  = threadIdx.x;
    const int lane = tid & 31;
    const int wave = tid >> 5;          // 0..7
    const int waveM = wave >> 1;        // 0..3 -> rows 32*waveM
    const int waveN = wave & 1;         // 0..1 -> cols 64*waveN
    const int m0 = blockIdx.x * 128;
    const int n0 = blockIdx.y * 128;

    const u16* Ag = A  + (size_t)blockIdx.z * aBS;
    const u16* Bg = Bt + (size_t)blockIdx.z * bBS;

    v8f acc[2][4];
#pragma unroll
    for (int i = 0; i < 2; ++i)
#pragma unroll
        for (int j = 0; j < 4; ++j)
            acc[i][j] = v8f{0.f, 0.f, 0.f, 0.f, 0.f, 0.f, 0.f, 0.f};

    const int nStages = Ktot >> 6;      // K staged 64 wide

    auto compute = [&](int cur) {
        const u16* lA = (const u16*)&lbuf[cur][0];
        const u16* lB = (const u16*)&lbuf[cur][1024];
#pragma unroll
        for (int kk = 0; kk < 2; ++kk) {
            const int kf = kk * 32;
            v16bf af[2], bfr[4];
#pragma unroll
            for (int i = 0; i < 2; ++i)
                af[i] = ldfragA(lA, 32 * waveM + 16 * i, kf, lane);
#pragma unroll
            for (int j = 0; j < 4; ++j)
                bfr[j] = ldfragB(lB, 64 * waveN + 16 * j, kf, lane);
#pragma unroll
            for (int i = 0; i < 2; ++i)
#pragma unroll
                for (int j = 0; j < 4; ++j)
                    acc[i][j] = __builtin_amdgcn_wmma_f32_16x16x32_bf16(
                        false, af[i], false, bfr[j], (short)0, acc[i][j],
                        false, false);
        }
    };

#if USE_ASYNC_LDS
    // per-thread source pointers, advanced 64 elements (128 B) per stage
    const u16* ap[4];
    const u16* bp[4];
#pragma unroll
    for (int rep = 0; rep < 4; ++rep) {
        const int idx = tid + rep * 256;          // 1024 uint4 per tile
        const int row = idx >> 3, q = idx & 7;
        ap[rep] = Ag + (size_t)(m0 + row) * lda + q * 8;
        bp[rep] = Bg + (size_t)(n0 + row) * ldb + q * 8;
    }
    auto prefetch = [&](int buf) {               // 8 async-LDS instr / wave
#pragma unroll
        for (int rep = 0; rep < 4; ++rep) {
            const int idx = tid + rep * 256;
            __builtin_amdgcn_global_load_async_to_lds_b128(
                (gv4i_p)ap[rep], (lv4i_p)&lbuf[buf][idx], 0, 0);
            __builtin_amdgcn_global_load_async_to_lds_b128(
                (gv4i_p)bp[rep], (lv4i_p)&lbuf[buf][1024 + idx], 0, 0);
            ap[rep] += 64;
            bp[rep] += 64;
        }
    };

    prefetch(0);
    int cur = 0;
    for (int s = 0; s < nStages - 1; ++s) {      // steady state: always prefetch
        prefetch(cur ^ 1);
        __builtin_amdgcn_s_wait_asynccnt(8);     // stage s landed; prefetch in flight
        __syncthreads();
        compute(cur);
        __syncthreads();                         // readers done before next overwrite
        cur ^= 1;
    }
    __builtin_amdgcn_s_wait_asynccnt(0);         // epilogue: last stage
    __syncthreads();
    compute(cur);
#else
    // ---- synchronous single-buffer fallback
    for (int s = 0; s < nStages; ++s) {
        const int k0 = s << 6;
        __syncthreads();
#pragma unroll
        for (int rep = 0; rep < 4; ++rep) {
            const int idx = tid + rep * 256;      // 1024 uint4 per tile
            const int row = idx >> 3, q = idx & 7;
            lbuf[0][idx]        = *((const uint4*)(Ag + (size_t)(m0 + row) * lda + k0) + q);
            lbuf[0][1024 + idx] = *((const uint4*)(Bg + (size_t)(n0 + row) * ldb + k0) + q);
        }
        __syncthreads();
        compute(0);
    }
#endif

    // ---- store: D-frag layout: col n = lane&15; row m = g + 8*(lane>>4) ----
    const int cn = lane & 15;
    const int hi = lane >> 4;
    if constexpr (MODE == 0) {
        u16* Cc = (u16*)C + (size_t)blockIdx.z * cBS;
#pragma unroll
        for (int i = 0; i < 2; ++i)
#pragma unroll
            for (int j = 0; j < 4; ++j) {
                const int n  = n0 + 64 * waveN + 16 * j + cn;
                const int mb = m0 + 32 * waveM + 16 * i + 8 * hi;
#pragma unroll
                for (int g = 0; g < 8; ++g)
                    Cc[(size_t)(mb + g) * ldc + n] = f2bf(acc[i][j][g]);
            }
    } else if constexpr (MODE == 1) {
        u16* Cc = (u16*)C + (size_t)blockIdx.z * cBS;
#pragma unroll
        for (int i = 0; i < 2; ++i)
#pragma unroll
            for (int j = 0; j < 4; ++j) {
                const int n  = n0 + 64 * waveN + 16 * j + cn;
                const int mb = m0 + 32 * waveM + 16 * i + 8 * hi;
                union { uint4 q; u16 h[8]; } u;
#pragma unroll
                for (int g = 0; g < 8; ++g) u.h[g] = f2bf(acc[i][j][g]);
                *(uint4*)(Cc + (size_t)n * ldc + mb) = u.q;   // packed b128
            }
    } else {
        float* Cf = (float*)C + (size_t)blockIdx.z * cBS;
#pragma unroll
        for (int i = 0; i < 2; ++i)
#pragma unroll
            for (int j = 0; j < 4; ++j) {
                const int n  = n0 + 64 * waveN + 16 * j + cn;
                const int mb = m0 + 32 * waveM + 16 * i + 8 * hi;
#pragma unroll
                for (int g = 0; g < 8; ++g)
                    Cf[(size_t)(mb + g) * ldc + n] = acc[i][j][g];
            }
    }
}

// ---------------------------------------------------------------------------
// In-place row softmax over bf16 scores. One 256-thread block per row (n=2048,
// 8 elements/thread). Scale folded in (1/sqrt(d_out) = 1/32).
// ---------------------------------------------------------------------------
__global__ void softmax_rows(u16* __restrict__ S, int n, float scale) {
    __shared__ float red[8];
    const int tid  = threadIdx.x;
    const int lane = tid & 31;
    const int wave = tid >> 5;
    u16* row = S + (size_t)blockIdx.x * n + tid * 8;

    union { uint4 q; u16 h[8]; } u;
    u.q = *(const uint4*)row;
    float v[8];
#pragma unroll
    for (int i = 0; i < 8; ++i) v[i] = bf2f(u.h[i]) * scale;

    float m = v[0];
#pragma unroll
    for (int i = 1; i < 8; ++i) m = fmaxf(m, v[i]);
#pragma unroll
    for (int o = 16; o > 0; o >>= 1) m = fmaxf(m, __shfl_xor(m, o, 32));
    if (lane == 0) red[wave] = m;
    __syncthreads();
    float M = red[0];
#pragma unroll
    for (int w = 1; w < 8; ++w) M = fmaxf(M, red[w]);
    __syncthreads();

    float e[8], s = 0.f;
#pragma unroll
    for (int i = 0; i < 8; ++i) { e[i] = __expf(v[i] - M); s += e[i]; }
#pragma unroll
    for (int o = 16; o > 0; o >>= 1) s += __shfl_xor(s, o, 32);
    if (lane == 0) red[wave] = s;
    __syncthreads();
    float T = 0.f;
#pragma unroll
    for (int w = 0; w < 8; ++w) T += red[w];
    const float inv = 1.0f / T;
#pragma unroll
    for (int i = 0; i < 8; ++i) u.h[i] = f2bf(e[i] * inv);
    *(uint4*)row = u.q;
}

// ---------------------------------------------------------------------------
// Host-side orchestration
// ---------------------------------------------------------------------------
extern "C" void kernel_launch(void* const* d_in, const int* in_sizes, int n_in,
                              void* d_out, int out_size, void* d_ws, size_t ws_size,
                              hipStream_t stream) {
    (void)in_sizes; (void)n_in; (void)out_size; (void)ws_size;
    const float* x  = (const float*)d_in[0];   // [8,2048,1024]
    const float* wk = (const float*)d_in[1];   // [3,1024,1024]
    float* out = (float*)d_out;                // [8,2048,1024] f32

    const size_t M  = 16384;                   // B*S
    const size_t D  = 1024;
    const size_t S  = 2048;
    const size_t Bn = 8;

    u16* xb  = (u16*)d_ws;                     // [M][D]            32 MB
    u16* wbT = xb  + M * D;                    // [3][D][D] (e,d)    6 MB
    u16* Qb  = wbT + 3 * D * D;                // [M][D]            32 MB
    u16* Kb  = Qb  + M * D;                    // [M][D]            32 MB
    u16* Vt  = Kb  + M * D;                    // [D][M] transposed 32 MB
    u16* Sc  = Vt  + M * D;                    // [B][S][S]         64 MB

    // 1. precision conversion (+ weight transpose)
    conv_f32_to_bf16<<<dim3((unsigned)(M * D / (8 * 256))), dim3(256), 0, stream>>>(x, xb);
    conv_transpose_w<<<dim3(32, 32, 3), dim3(32, 8), 0, stream>>>(wk, wbT);

    // 2. projections: Q, K row-major; V stored transposed as Vt[e][m]
    gemm_bf16_wmma<0><<<dim3(128, 8, 1), dim3(256), 0, stream>>>(
        xb, 0, (int)D, wbT,             0, (int)D, Qb, 0, (int)D, (int)D);
    gemm_bf16_wmma<0><<<dim3(128, 8, 1), dim3(256), 0, stream>>>(
        xb, 0, (int)D, wbT + 1 * D * D, 0, (int)D, Kb, 0, (int)D, (int)D);
    gemm_bf16_wmma<1><<<dim3(128, 8, 1), dim3(256), 0, stream>>>(
        xb, 0, (int)D, wbT + 2 * D * D, 0, (int)D, Vt, 0, (int)M, (int)D);

    // 3. scores = Q @ K^T per batch (K rows act as pre-transposed B operand)
    gemm_bf16_wmma<0><<<dim3(16, 16, (unsigned)Bn), dim3(256), 0, stream>>>(
        Qb, (long long)(S * D), (int)D,
        Kb, (long long)(S * D), (int)D,
        Sc, (long long)(S * S), (int)S, (int)D);

    // 4. softmax rows (scale = 1/sqrt(1024))
    softmax_rows<<<dim3((unsigned)(Bn * S)), dim3(256), 0, stream>>>(Sc, (int)S, 0.03125f);

    // 5. out = P @ V per batch, via Vt (Bt row e holds V[:,e] contiguously)
    gemm_bf16_wmma<2><<<dim3(16, 8, (unsigned)Bn), dim3(256), 0, stream>>>(
        Sc, (long long)(S * S), (int)S,
        Vt, (long long)S,       (int)M,
        out, (long long)(S * D), (int)D, (int)S);
}
